// SerializedAttention_71957882077526
// MI455X (gfx1250) — matrix operand order repaired and checked by reading, not compiled, not run on hardware
//
#include <hip/hip_runtime.h>
#include <hip/hip_bf16.h>

// ---------------- problem constants ----------------
#define NPTS  32768
#define CCH   512
#define NH    8
#define HD    64     // head dim
#define PK    128    // patch size
#define NP    256    // patches
#define RPE_ROWS 99  // 3*33
#define QSCALE 0.125f // 64^-0.5

typedef __attribute__((ext_vector_type(16))) __bf16 v16bf;
typedef __attribute__((ext_vector_type(8)))  __bf16 v8bf;
typedef __attribute__((ext_vector_type(8)))  float  v8f;
typedef __attribute__((ext_vector_type(4)))  int    v4i;

#define AS1 __attribute__((address_space(1)))
#define AS3 __attribute__((address_space(3)))

// gfx1250 async global->LDS copy path (guarded: falls back to sync copies if the
// toolchain doesn't declare the builtins).
#if defined(__has_builtin)
#if __has_builtin(__builtin_amdgcn_global_load_async_to_lds_b128) && \
    __has_builtin(__builtin_amdgcn_s_wait_asynccnt)
#define USE_ASYNC_LDS 1
#endif
#endif

static __device__ __forceinline__ __bf16 f2bf(float f) {
    unsigned u = __builtin_bit_cast(unsigned, f);
    u += 0x7fffu + ((u >> 16) & 1u);                 // round to nearest even
    unsigned short h = (unsigned short)(u >> 16);
    return __builtin_bit_cast(__bf16, h);
}

static __device__ __forceinline__ v16bf cat8(v8bf lo, v8bf hi) {
    return __builtin_shufflevector(lo, hi, 0,1,2,3,4,5,6,7,8,9,10,11,12,13,14,15);
}

// A fragment (16x32 bf16, M x K). base points at [mtile_row0][kb]. ld in elements.
// ISA: lanes 0-15 row M=lane hold K {0..7, 16..23}; lanes 16-31 row M=lane-16 hold K {8..15, 24..31}.
static __device__ __forceinline__ v16bf load_a_frag(const __bf16* base, int ld, int lane) {
    int half = lane >> 4;
    int r    = lane & 15;
    const __bf16* p = base + r * ld + half * 8;
    v8bf lo = *(const v8bf*)(p);
    v8bf hi = *(const v8bf*)(p + 16);
    return cat8(lo, hi);
}

// B fragment (32x16 bf16, K x N) read from an N-major buffer Bt[n][k]. base at [ntile_col0][kb].
// lane%16 = N column; lanes 0-15 hold K {0..7, 16..23}, lanes 16-31 hold K {8..15, 24..31}.
static __device__ __forceinline__ v16bf load_b_frag(const __bf16* base, int ld, int lane) {
    int half = lane >> 4;
    int nn   = lane & 15;
    const __bf16* p = base + nn * ld + half * 8;
    v8bf lo = *(const v8bf*)(p);
    v8bf hi = *(const v8bf*)(p + 16);
    return cat8(lo, hi);
}

static __device__ __forceinline__ v8f wmma_bf16(v16bf a, v16bf b, v8f c) {
    return __builtin_amdgcn_wmma_f32_16x16x32_bf16(false, a, false, b, (short)0, c, false, false);
}

// Stage a 128x32 bf16 tile (source leading dim 512) into a [128][40]-padded LDS tile.
// 512 chunks of 16B, 2 per thread. Async path issues ASYNCcnt-tracked DMA.
static __device__ __forceinline__ void stage_tile(const __bf16* __restrict__ g,
                                                  __bf16* l, int tid) {
    #pragma unroll
    for (int c = tid; c < 512; c += 256) {
        int row = c >> 2, q = (c & 3) << 3;
        const __bf16* src = g + row * 512 + q;
        __bf16* dst = l + row * 40 + q;
#if USE_ASYNC_LDS
        __builtin_amdgcn_global_load_async_to_lds_b128(
            (AS1 v4i*)(const void*)src, (AS3 v4i*)dst, 0, 0);
#else
        *(v8bf*)dst = *(const v8bf*)src;
#endif
    }
}

static __device__ __forceinline__ void async_tile_wait() {
#if USE_ASYNC_LDS
    __builtin_amdgcn_s_wait_asynccnt(0);
#endif
}

// ---------------- prep kernels ----------------
__global__ void prep_gather_feat(const float* __restrict__ feat,
                                 const int* __restrict__ order,
                                 __bf16* __restrict__ featg) {
    for (int idx = blockIdx.x * blockDim.x + threadIdx.x; idx < NPTS * CCH;
         idx += gridDim.x * blockDim.x) {
        int s = idx >> 9, c = idx & 511;
        featg[idx] = f2bf(feat[order[s] * CCH + c]);
    }
}

// wt[n][k] = bf16(w[k][n]); w is [rows=k][cols=n] row-major.
__global__ void prep_transpose_w(const float* __restrict__ w, __bf16* __restrict__ wt,
                                 int rows, int cols) {
    for (int idx = blockIdx.x * blockDim.x + threadIdx.x; idx < rows * cols;
         idx += gridDim.x * blockDim.x) {
        int n = idx / rows, k = idx - n * rows;
        wt[idx] = f2bf(w[k * cols + n]);
    }
}

__global__ void prep_gather_gc(const int* __restrict__ gc, const int* __restrict__ order,
                               int* __restrict__ gcs) {
    for (int idx = blockIdx.x * blockDim.x + threadIdx.x; idx < NPTS * 3;
         idx += gridDim.x * blockDim.x) {
        int s = idx / 3, d = idx - s * 3;
        gcs[idx] = gc[order[s] * 3 + d];
    }
}

// ---------------- QKV GEMM: [32768,512] x [512,1536], epilogue splits q/k/vT ----------------
__global__ __launch_bounds__(256) void qkv_gemm(
    const __bf16* __restrict__ Ag,    // featg bf16 [N, 512] (gathered by order)
    const __bf16* __restrict__ WT,    // w_qkv^T bf16 [1536][512]
    const float*  __restrict__ bqkv,  // [1536]
    __bf16* __restrict__ qbuf,        // [NP,NH,PK,HD] (scaled)
    __bf16* __restrict__ kbuf,        // [NP,NH,PK,HD]
    __bf16* __restrict__ vtbuf)       // [NP,NH,HD,PK]
{
    __shared__ __bf16 As[2][128 * 40];
    __shared__ __bf16 Bs[2][128 * 40];

    const int tid = threadIdx.x, lane = tid & 31, wave = tid >> 5;
    const int wm = wave & 1, wn = wave >> 1;
    const int rowBase = blockIdx.x * 128;   // 256 blocks
    const int colBase = blockIdx.y * 128;   // 12 blocks

    const __bf16* Abase = Ag + (size_t)rowBase * CCH;
    const __bf16* Bbase = WT + (size_t)colBase * CCH;

    v8f acc[4][2];
    const v8f vzero = {0.f,0.f,0.f,0.f,0.f,0.f,0.f,0.f};
    #pragma unroll
    for (int mt = 0; mt < 4; ++mt)
        #pragma unroll
        for (int nt = 0; nt < 2; ++nt) acc[mt][nt] = vzero;

    stage_tile(Abase, As[0], tid);
    stage_tile(Bbase, Bs[0], tid);

    int buf = 0;
    for (int kb = 0; kb < CCH; kb += 32) {
        async_tile_wait();
        __syncthreads();
        if (kb + 32 < CCH) {                 // prefetch next tile into other buffer
            stage_tile(Abase + kb + 32, As[buf ^ 1], tid);
            stage_tile(Bbase + kb + 32, Bs[buf ^ 1], tid);
        }
        v16bf bfr[2];
        #pragma unroll
        for (int nt = 0; nt < 2; ++nt)
            bfr[nt] = load_b_frag(&Bs[buf][(wn * 32 + nt * 16) * 40], 40, lane);
        #pragma unroll
        for (int mt = 0; mt < 4; ++mt) {
            v16bf af = load_a_frag(&As[buf][(wm * 64 + mt * 16) * 40], 40, lane);
            #pragma unroll
            for (int nt = 0; nt < 2; ++nt)
                acc[mt][nt] = wmma_bf16(af, bfr[nt], acc[mt][nt]);
        }
        buf ^= 1;
    }

    // epilogue: patch == blockIdx.x since M-tile == patch size
    const int p = blockIdx.x;
    const int half8 = (lane >> 4) << 3;
    #pragma unroll
    for (int mt = 0; mt < 4; ++mt)
    #pragma unroll
    for (int nt = 0; nt < 2; ++nt)
    #pragma unroll
    for (int r = 0; r < 8; ++r) {
        int ki = wm * 64 + mt * 16 + r + half8;               // row within patch
        int cg = colBase + wn * 32 + nt * 16 + (lane & 15);   // col in [0,1536)
        float val = acc[mt][nt][r] + bqkv[cg];
        int which = cg >> 9;
        int h  = (cg >> 6) & 7;
        int dd = cg & 63;
        size_t phk = ((size_t)(p * NH + h) * PK + ki);
        if (which == 0)       qbuf[phk * HD + dd] = f2bf(val * QSCALE);
        else if (which == 1)  kbuf[phk * HD + dd] = f2bf(val);
        else                  vtbuf[((size_t)(p * NH + h) * HD + dd) * PK + ki] = f2bf(val);
    }
}

// ---------------- attention: one block per (patch, head) ----------------
__global__ __launch_bounds__(256) void attn_kernel(
    const __bf16* __restrict__ qbuf,
    const __bf16* __restrict__ kbuf,
    const __bf16* __restrict__ vtbuf,
    const int*   __restrict__ gcs,    // [N,3] serialized order
    const float* __restrict__ rpe,    // [99, 8]
    __bf16* __restrict__ aout)        // [N, 512] serialized
{
    __shared__ __bf16 Ps[128 * 136];
    __shared__ int    gcsh[128 * 3];
    __shared__ float  rpes[RPE_ROWS];

    const int tid = threadIdx.x, lane = tid & 31, wave = tid >> 5;
    const int ph = blockIdx.x, p = ph >> 3, h = ph & 7;

    for (int idx = tid; idx < 384; idx += 256) gcsh[idx] = gcs[p * 384 + idx];
    if (tid < RPE_ROWS) rpes[tid] = rpe[tid * NH + h];
    __syncthreads();

    const __bf16* Q  = qbuf  + (size_t)ph * PK * HD;
    const __bf16* Kt = kbuf  + (size_t)ph * PK * HD;
    const __bf16* Vt = vtbuf + (size_t)ph * HD * PK;

    const v8f vzero = {0.f,0.f,0.f,0.f,0.f,0.f,0.f,0.f};
    v8f s[8];
    #pragma unroll
    for (int t = 0; t < 8; ++t) s[t] = vzero;

    // S = (q*scale) k^T : M=i (16 rows per wave), N=j (8 tiles), K=d (2 steps)
    #pragma unroll
    for (int kb = 0; kb < HD; kb += 32) {
        v16bf aq = load_a_frag(Q + (size_t)(wave * 16) * HD + kb, HD, lane);
        #pragma unroll
        for (int t = 0; t < 8; ++t) {
            v16bf bk = load_b_frag(Kt + (size_t)(t * 16) * HD + kb, HD, lane);
            s[t] = wmma_bf16(aq, bk, s[t]);
        }
    }

    // RPE bias + softmax entirely in registers (row lives in one 16-lane half)
    const int half8 = (lane >> 4) << 3;
    const int jc    = lane & 15;
    #pragma unroll
    for (int r = 0; r < 8; ++r) {
        int i = wave * 16 + r + half8;
        int gix = gcsh[i * 3 + 0], giy = gcsh[i * 3 + 1], giz = gcsh[i * 3 + 2];
        float vals[8];
        float m = -3.0e38f;
        #pragma unroll
        for (int t = 0; t < 8; ++t) {
            int j = t * 16 + jc;
            int rx = gix - gcsh[j * 3 + 0];
            int ry = giy - gcsh[j * 3 + 1];
            int rz = giz - gcsh[j * 3 + 2];
            rx = rx < -16 ? -16 : (rx > 16 ? 16 : rx);
            ry = ry < -16 ? -16 : (ry > 16 ? 16 : ry);
            rz = rz < -16 ? -16 : (rz > 16 ? 16 : rz);
            float b3 = rpes[rx + 16] + rpes[ry + 16 + 33] + rpes[rz + 16 + 66];
            vals[t] = s[t][r] + b3;
            m = fmaxf(m, vals[t]);
        }
        m = fmaxf(m, __shfl_xor(m, 1, 16));
        m = fmaxf(m, __shfl_xor(m, 2, 16));
        m = fmaxf(m, __shfl_xor(m, 4, 16));
        m = fmaxf(m, __shfl_xor(m, 8, 16));
        float sum = 0.f;
        #pragma unroll
        for (int t = 0; t < 8; ++t) { vals[t] = __expf(vals[t] - m); sum += vals[t]; }
        sum += __shfl_xor(sum, 1, 16);
        sum += __shfl_xor(sum, 2, 16);
        sum += __shfl_xor(sum, 4, 16);
        sum += __shfl_xor(sum, 8, 16);
        float inv = 1.0f / sum;
        #pragma unroll
        for (int t = 0; t < 8; ++t)
            Ps[i * 136 + t * 16 + jc] = f2bf(vals[t] * inv);
    }
    __syncthreads();

    // O = P v : M=i, N=d (4 tiles), K=j (4 steps of 32); v^T rows give contiguous B frags
    v8f o[4];
    #pragma unroll
    for (int t = 0; t < 4; ++t) o[t] = vzero;
    #pragma unroll
    for (int jb = 0; jb < PK; jb += 32) {
        v16bf ap = load_a_frag(&Ps[(wave * 16) * 136 + jb], 136, lane);
        #pragma unroll
        for (int t = 0; t < 4; ++t) {
            v16bf bv = load_b_frag(Vt + (size_t)(t * 16) * PK + jb, PK, lane);
            o[t] = wmma_bf16(ap, bv, o[t]);
        }
    }

    #pragma unroll
    for (int t = 0; t < 4; ++t)
    #pragma unroll
    for (int r = 0; r < 8; ++r) {
        int i  = wave * 16 + r + half8;
        int dd = t * 16 + jc;
        aout[(size_t)(p * PK + i) * CCH + h * HD + dd] = f2bf(o[t][r]);
    }
}

// ---------------- projection GEMM + bias + residual + scatter to original order ----------------
__global__ __launch_bounds__(256) void proj_gemm(
    const __bf16* __restrict__ Ag,     // aout bf16 [N, 512] serialized
    const __bf16* __restrict__ WT,     // w_proj^T bf16 [512][512]
    const float*  __restrict__ bproj,  // [512]
    const int*    __restrict__ order,  // [N]
    const float*  __restrict__ feat,   // [N, 512] original order
    float* __restrict__ out)           // [N, 512] original order
{
    __shared__ __bf16 As[2][128 * 40];
    __shared__ __bf16 Bs[2][128 * 40];

    const int tid = threadIdx.x, lane = tid & 31, wave = tid >> 5;
    const int wm = wave & 1, wn = wave >> 1;
    const int rowBase = blockIdx.x * 128;   // 256 blocks (serialized rows)
    const int colBase = blockIdx.y * 128;   // 4 blocks

    const __bf16* Abase = Ag + (size_t)rowBase * CCH;
    const __bf16* Bbase = WT + (size_t)colBase * CCH;

    v8f acc[4][2];
    const v8f vzero = {0.f,0.f,0.f,0.f,0.f,0.f,0.f,0.f};
    #pragma unroll
    for (int mt = 0; mt < 4; ++mt)
        #pragma unroll
        for (int nt = 0; nt < 2; ++nt) acc[mt][nt] = vzero;

    stage_tile(Abase, As[0], tid);
    stage_tile(Bbase, Bs[0], tid);

    int buf = 0;
    for (int kb = 0; kb < CCH; kb += 32) {
        async_tile_wait();
        __syncthreads();
        if (kb + 32 < CCH) {
            stage_tile(Abase + kb + 32, As[buf ^ 1], tid);
            stage_tile(Bbase + kb + 32, Bs[buf ^ 1], tid);
        }
        v16bf bfr[2];
        #pragma unroll
        for (int nt = 0; nt < 2; ++nt)
            bfr[nt] = load_b_frag(&Bs[buf][(wn * 32 + nt * 16) * 40], 40, lane);
        #pragma unroll
        for (int mt = 0; mt < 4; ++mt) {
            v16bf af = load_a_frag(&As[buf][(wm * 64 + mt * 16) * 40], 40, lane);
            #pragma unroll
            for (int nt = 0; nt < 2; ++nt)
                acc[mt][nt] = wmma_bf16(af, bfr[nt], acc[mt][nt]);
        }
        buf ^= 1;
    }

    const int half8 = (lane >> 4) << 3;
    #pragma unroll
    for (int mt = 0; mt < 4; ++mt)
    #pragma unroll
    for (int nt = 0; nt < 2; ++nt)
    #pragma unroll
    for (int r = 0; r < 8; ++r) {
        int srow = rowBase + wm * 64 + mt * 16 + r + half8;   // serialized row
        int n    = colBase + wn * 32 + nt * 16 + (lane & 15);
        int dest = order[srow];                                // original point index
        float v = acc[mt][nt][r] + bproj[n];
        out[(size_t)dest * CCH + n] = feat[(size_t)dest * CCH + n] + v;
    }
}

// ---------------- host-side launch ----------------
extern "C" void kernel_launch(void* const* d_in, const int* in_sizes, int n_in,
                              void* d_out, int out_size, void* d_ws, size_t ws_size,
                              hipStream_t stream) {
    const float* feat   = (const float*)d_in[0];
    const int*   gcoord = (const int*)  d_in[1];
    const int*   order  = (const int*)  d_in[2];
    // d_in[3] = inverse (not needed: scatter via order in proj epilogue)
    const float* w_qkv  = (const float*)d_in[4];
    const float* b_qkv  = (const float*)d_in[5];
    const float* w_proj = (const float*)d_in[6];
    const float* b_proj = (const float*)d_in[7];
    const float* rpe    = (const float*)d_in[8];
    float* out = (float*)d_out;

    char* ws = (char*)d_ws;
    size_t o = 0;
    auto take = [&](size_t bytes) -> char* {
        char* pp = ws + o;
        o = (o + bytes + 255) & ~(size_t)255;
        return pp;
    };
    __bf16* featg  = (__bf16*)take((size_t)NPTS * CCH * 2);        // 32 MB
    __bf16* wqkvT  = (__bf16*)take((size_t)CCH * 1536 * 2);        // 1.5 MB [1536][512]
    __bf16* wprojT = (__bf16*)take((size_t)CCH * CCH * 2);         // 0.5 MB [512][512]
    int*    gcsd   = (int*)   take((size_t)NPTS * 3 * 4);          // 0.4 MB
    __bf16* qb     = (__bf16*)take((size_t)NP * NH * PK * HD * 2); // 32 MB
    __bf16* kb     = (__bf16*)take((size_t)NP * NH * PK * HD * 2); // 32 MB
    __bf16* vtb    = (__bf16*)take((size_t)NP * NH * HD * PK * 2); // 32 MB
    __bf16* aoutb  = (__bf16*)take((size_t)NPTS * CCH * 2);        // 32 MB
    (void)in_sizes; (void)n_in; (void)out_size; (void)ws_size;

    // prep: gather feat (bf16), transpose+convert weights, gather grid coords
    prep_gather_feat<<<4096, 256, 0, stream>>>(feat, order, featg);
    prep_transpose_w<<<768, 256, 0, stream>>>(w_qkv, wqkvT, CCH, 1536);
    prep_transpose_w<<<256, 256, 0, stream>>>(w_proj, wprojT, CCH, CCH);
    prep_gather_gc<<<96, 256, 0, stream>>>(gcoord, order, gcsd);

    // QKV GEMM -> q (scaled), k, v^T staging buffers
    qkv_gemm<<<dim3(NP, 12), 256, 0, stream>>>(featg, wqkvT, b_qkv, qb, kb, vtb);

    // attention per (patch, head)
    attn_kernel<<<NP * NH, 256, 0, stream>>>(qb, kb, vtb, gcsd, rpe, aoutb);

    // projection + residual + scatter back to original order
    proj_gemm<<<dim3(NP, 4), 256, 0, stream>>>(aoutb, wprojT, b_proj, order, feat, out);
}
